// GPT2Attention_22239340659491
// MI455X (gfx1250) — compile-verified, matrix operand order checked by reading, not run on hardware
//
#include <hip/hip_runtime.h>
#include <hip/hip_bf16.h>
#include <math.h>

typedef __bf16 bf16;
typedef __bf16 v16bf __attribute__((ext_vector_type(16)));
typedef float  v8f   __attribute__((ext_vector_type(8)));
typedef int    v4i   __attribute__((vector_size(16)));   // matches builtin param

#define B_  2
#define S_  2048
#define D_  1024
#define H_  16
#define HD_ 64
#define TK  64

#define GLOBAL_AS __attribute__((address_space(1)))
#define LDS_AS    __attribute__((address_space(3)))

#if defined(__has_builtin)
#if __has_builtin(__builtin_amdgcn_global_load_async_to_lds_b128)
#define HAS_ASYNC_LDS 1
#endif
#endif

#if defined(HAS_ASYNC_LDS)
#define ASYNC_CP16(g, l)                                                        \
  __builtin_amdgcn_global_load_async_to_lds_b128(                               \
      (GLOBAL_AS v4i*)(g), (LDS_AS v4i*)(l), 0, 0)
#if __has_builtin(__builtin_amdgcn_s_wait_asynccnt)
#define WAIT_ASYNC(n) __builtin_amdgcn_s_wait_asynccnt(n)
#else
#define WAIT_ASYNC(n) asm volatile("s_wait_asynccnt %0" ::"i"(n) : "memory")
#endif
#else
// Fallback: synchronous VGPR round-trip copy
#define ASYNC_CP16(g, l) (*(uint4*)(l) = *(const uint4*)(g))
#define WAIT_ASYNC(n) ((void)0)
#endif

// Load one 16x32 bf16 WMMA fragment (A-layout; identical B-layout when the
// tile is stored [N][K]).
// lanes 0-15 : row = lane,    K chunks [0..7]   and [16..23]
// lanes 16-31: row = lane-16, K chunks [8..15]  and [24..31]
__device__ inline v16bf load_frag16(const bf16* base, int ldr, int lane) {
  const int r  = lane & 15;
  const int kf = ((lane >> 4) & 1) * 8;
  union { v16bf v; uint4 u[2]; } f;
  f.u[0] = *(const uint4*)(base + (size_t)r * ldr + kf);
  f.u[1] = *(const uint4*)(base + (size_t)r * ldr + kf + 16);
  return f.v;
}

__global__ __launch_bounds__(256) void cvt_f32_to_bf16(
    const float* __restrict__ in, bf16* __restrict__ out, int n) {
  for (int i = blockIdx.x * blockDim.x + threadIdx.x; i < n;
       i += gridDim.x * blockDim.x)
    out[i] = (bf16)in[i];
}

// W [K][N] f32  ->  Wt [N][K] bf16 (one-time; makes GEMM B-tiles contiguous)
__global__ __launch_bounds__(256) void cvt_transpose_bf16(
    const float* __restrict__ in, bf16* __restrict__ out, int K, int N) {
  for (int i = blockIdx.x * blockDim.x + threadIdx.x; i < K * N;
       i += gridDim.x * blockDim.x) {
    int k = i / N, n = i - k * N;
    out[(size_t)n * K + k] = (bf16)in[i];
  }
}

// C[M,N] = A[M,K] * Bt[N,K]^T, bf16 in, f32 accumulate.
// Double-buffered async global->LDS staging, K-tile = 64 (2 WMMA K-chunks).
// mode 0: f32 C row-major.  mode 1: QKV scatter epilogue.
__global__ __launch_bounds__(256) void gemm_wmma_bf16(
    const bf16* __restrict__ A, const bf16* __restrict__ Bt,
    int M, int N, int Kd, int mode,
    float* __restrict__ Cf, bf16* __restrict__ Qb,
    bf16* __restrict__ Kb, bf16* __restrict__ Vtb) {
  __shared__ bf16 sA[2][128 * TK];   // [m][k]
  __shared__ bf16 sB[2][128 * TK];   // [n][k]
  const int tid  = threadIdx.x;
  const int lane = tid & 31;
  const int w    = tid >> 5;
  const int wm   = w >> 1;        // 0..3 (M)
  const int wn   = w & 1;         // 0..1 (N)
  const int m0   = blockIdx.y * 128;
  const int n0   = blockIdx.x * 128;

  // Each thread stages 4 x 16B chunks per operand per K-stage.
  auto stage = [&](int buf, int k0) {
#pragma unroll
    for (int i = 0; i < 4; ++i) {
      int idx = tid * 4 + i;          // 1024 chunks of 8 bf16 = 128x64 tile
      int row = idx >> 3;
      int c   = (idx & 7) * 8;
      ASYNC_CP16(&A[(size_t)(m0 + row) * Kd + k0 + c], &sA[buf][row * TK + c]);
      ASYNC_CP16(&Bt[(size_t)(n0 + row) * Kd + k0 + c], &sB[buf][row * TK + c]);
    }
  };

  const v8f vzero = {0.f,0.f,0.f,0.f,0.f,0.f,0.f,0.f};
  v8f acc[2][4];
#pragma unroll
  for (int mi = 0; mi < 2; ++mi)
#pragma unroll
    for (int ni = 0; ni < 4; ++ni) acc[mi][ni] = vzero;

  stage(0, 0);
  const int nk = Kd / TK;
  for (int t = 0; t < nk; ++t) {
    const int cur = t & 1;
    if (t + 1 < nk) {
      stage(cur ^ 1, (t + 1) * TK);   // prefetch next stage into other buffer
      WAIT_ASYNC(8);                  // oldest 8 ops (current stage) complete
    } else {
      WAIT_ASYNC(0);
    }
    __syncthreads();
#pragma unroll
    for (int kc = 0; kc < 2; ++kc) {
      v16bf af[2], bfr[4];
#pragma unroll
      for (int mi = 0; mi < 2; ++mi)
        af[mi] = load_frag16(&sA[cur][(wm * 32 + mi * 16) * TK + kc * 32], TK, lane);
#pragma unroll
      for (int ni = 0; ni < 4; ++ni)
        bfr[ni] = load_frag16(&sB[cur][(wn * 64 + ni * 16) * TK + kc * 32], TK, lane);
#pragma unroll
      for (int mi = 0; mi < 2; ++mi)
#pragma unroll
        for (int ni = 0; ni < 4; ++ni)
          acc[mi][ni] = __builtin_amdgcn_wmma_f32_16x16x32_bf16(
              false, af[mi], false, bfr[ni], (short)0, acc[mi][ni], false, false);
    }
    __syncthreads();                  // all reads done before next overwrite
  }

  const int hi8 = (lane >> 4) * 8;
  const int cn  = lane & 15;
#pragma unroll
  for (int mi = 0; mi < 2; ++mi)
#pragma unroll
    for (int ni = 0; ni < 4; ++ni)
#pragma unroll
      for (int r = 0; r < 8; ++r) {
        int m   = m0 + wm * 32 + mi * 16 + hi8 + r;
        int n   = n0 + wn * 64 + ni * 16 + cn;
        float v = acc[mi][ni][r];
        if (mode == 0) {
          Cf[(size_t)m * N + n] = v;
        } else {
          int b = m >> 11, s = m & (S_ - 1);
          if (n < D_) {
            int h = n >> 6, hd = n & 63;   // Q: fold 1/sqrt(HD)
            Qb[(((size_t)(b * H_ + h)) * S_ + s) * HD_ + hd] = (bf16)(v * 0.125f);
          } else if (n < 2 * D_) {
            int nn = n - D_;
            int h = nn >> 6, hd = nn & 63;
            Kb[(((size_t)(b * H_ + h)) * S_ + s) * HD_ + hd] = (bf16)v;
          } else {
            int nn = n - 2 * D_;
            int h = nn >> 6, hd = nn & 63; // V stored transposed [B,H,HD,S]
            Vtb[(((size_t)(b * H_ + h)) * HD_ + hd) * S_ + s] = (bf16)v;
          }
        }
      }
}

// Flash-style attention, no mask. 4 waves/block, each wave owns 32 queries.
__global__ __launch_bounds__(128) void attn_fwd(
    const bf16* __restrict__ Q, const bf16* __restrict__ K,
    const bf16* __restrict__ Vt, bf16* __restrict__ Aout) {
  __shared__ bf16 sP[4][32 * 64];
  const int tid  = threadIdx.x;
  const int lane = tid & 31;
  const int w    = tid >> 5;
  const int h    = blockIdx.y;
  const int b    = blockIdx.z;
  const int q0   = blockIdx.x * 128 + w * 32;
  const int hi8  = (lane >> 4) * 8;
  const int cn   = lane & 15;

  const bf16* Qh = Q  + ((size_t)(b * H_ + h)) * S_ * HD_;
  const bf16* Kh = K  + ((size_t)(b * H_ + h)) * S_ * HD_;
  const bf16* Vh = Vt + ((size_t)(b * H_ + h)) * HD_ * S_;

  v16bf qf[2][2];
#pragma unroll
  for (int mi = 0; mi < 2; ++mi)
#pragma unroll
    for (int kc = 0; kc < 2; ++kc)
      qf[mi][kc] = load_frag16(Qh + (size_t)(q0 + mi * 16) * HD_ + kc * 32, HD_, lane);

  const v8f vzero = {0.f,0.f,0.f,0.f,0.f,0.f,0.f,0.f};
  v8f oacc[2][4];
  float mrun[2][8], lrun[2][8];
#pragma unroll
  for (int mi = 0; mi < 2; ++mi) {
#pragma unroll
    for (int oj = 0; oj < 4; ++oj) oacc[mi][oj] = vzero;
#pragma unroll
    for (int r = 0; r < 8; ++r) { mrun[mi][r] = -1e30f; lrun[mi][r] = 0.f; }
  }

  for (int kb = 0; kb < S_ / 64; ++kb) {
    const int kbase = kb * 64;
    if (kb + 1 < S_ / 64) {          // hint next key/value tiles into cache
      __builtin_prefetch(Kh + (size_t)(kbase + 64 + (lane & 63)) * HD_, 0, 0);
      __builtin_prefetch(Vh + (size_t)(lane & 63) * S_ + kbase + 64, 0, 0);
    }
    // ---- scores S = (Q/8) * K^T, 32x64 tile in f32 accumulators ----
    v8f sacc[2][4];
#pragma unroll
    for (int mi = 0; mi < 2; ++mi)
#pragma unroll
      for (int nj = 0; nj < 4; ++nj) sacc[mi][nj] = vzero;
#pragma unroll
    for (int kc = 0; kc < 2; ++kc) {
      v16bf kf[4];
#pragma unroll
      for (int nj = 0; nj < 4; ++nj)
        kf[nj] = load_frag16(Kh + (size_t)(kbase + nj * 16) * HD_ + kc * 32, HD_, lane);
#pragma unroll
      for (int mi = 0; mi < 2; ++mi)
#pragma unroll
        for (int nj = 0; nj < 4; ++nj)
          sacc[mi][nj] = __builtin_amdgcn_wmma_f32_16x16x32_bf16(
              false, qf[mi][kc], false, kf[nj], (short)0, sacc[mi][nj], false, false);
    }
    // ---- online softmax (rows live in 16-lane halves; xor-reduce 1..8) ----
#pragma unroll
    for (int mi = 0; mi < 2; ++mi) {
#pragma unroll
      for (int r = 0; r < 8; ++r) {
        float rm = sacc[mi][0][r];
#pragma unroll
        for (int nj = 1; nj < 4; ++nj) rm = fmaxf(rm, sacc[mi][nj][r]);
#pragma unroll
        for (int xm = 1; xm <= 8; xm <<= 1) rm = fmaxf(rm, __shfl_xor(rm, xm, 32));
        float mnew = fmaxf(mrun[mi][r], rm);
        float corr = __expf(mrun[mi][r] - mnew);
        float rs = 0.f;
#pragma unroll
        for (int nj = 0; nj < 4; ++nj) {
          float p = __expf(sacc[mi][nj][r] - mnew);
          sacc[mi][nj][r] = p;
          rs += p;
        }
#pragma unroll
        for (int xm = 1; xm <= 8; xm <<= 1) rs += __shfl_xor(rs, xm, 32);
        lrun[mi][r] = lrun[mi][r] * corr + rs;
        mrun[mi][r] = mnew;
#pragma unroll
        for (int oj = 0; oj < 4; ++oj) oacc[mi][oj][r] *= corr;
      }
      // P: D-layout -> row-major bf16 in this wave's private LDS tile
#pragma unroll
      for (int nj = 0; nj < 4; ++nj)
#pragma unroll
        for (int r = 0; r < 8; ++r)
          sP[w][(mi * 16 + hi8 + r) * 64 + nj * 16 + cn] = (bf16)sacc[mi][nj][r];
    }
    asm volatile("s_wait_dscnt 0" ::: "memory");  // within-wave LDS RAW
    // ---- O += P * V ----
#pragma unroll
    for (int kc = 0; kc < 2; ++kc) {
      v16bf pf[2], vf[4];
#pragma unroll
      for (int mi = 0; mi < 2; ++mi)
        pf[mi] = load_frag16(&sP[w][(mi * 16) * 64 + kc * 32], 64, lane);
#pragma unroll
      for (int oj = 0; oj < 4; ++oj)
        vf[oj] = load_frag16(Vh + (size_t)(oj * 16) * S_ + kbase + kc * 32, S_, lane);
#pragma unroll
      for (int mi = 0; mi < 2; ++mi)
#pragma unroll
        for (int oj = 0; oj < 4; ++oj)
          oacc[mi][oj] = __builtin_amdgcn_wmma_f32_16x16x32_bf16(
              false, pf[mi], false, vf[oj], (short)0, oacc[mi][oj], false, false);
    }
  }
  // ---- normalize and store merged-head bf16 [B,S,D] ----
#pragma unroll
  for (int mi = 0; mi < 2; ++mi)
#pragma unroll
    for (int oj = 0; oj < 4; ++oj)
#pragma unroll
      for (int r = 0; r < 8; ++r) {
        int s   = q0 + mi * 16 + hi8 + r;
        int col = oj * 16 + cn;
        float v = oacc[mi][oj][r] / lrun[mi][r];
        Aout[((size_t)(b * S_ + s)) * D_ + h * HD_ + col] = (bf16)v;
      }
}

extern "C" void kernel_launch(void* const* d_in, const int* in_sizes, int n_in,
                              void* d_out, int out_size, void* d_ws, size_t ws_size,
                              hipStream_t stream) {
  const float* hs = (const float*)d_in[0];   // [B,S,D]
  const float* wa = (const float*)d_in[1];   // [D,3D]
  const float* wp = (const float*)d_in[2];   // [D,D]
  float* out = (float*)d_out;

  char* ws = (char*)d_ws;
  bf16* Xb  = (bf16*)(ws);                          // 8 MiB  [4096,1024]
  bf16* Wat = (bf16*)(ws + ((size_t)8  << 20));     // 6 MiB  [3072,1024] = W_attn^T
  bf16* Wpt = (bf16*)(ws + ((size_t)14 << 20));     // 2 MiB  [1024,1024] = W_proj^T
  bf16* Qb  = (bf16*)(ws + ((size_t)16 << 20));     // 8 MiB  [B,H,S,HD] (x0.125)
  bf16* Kb  = (bf16*)(ws + ((size_t)24 << 20));     // 8 MiB  [B,H,S,HD]
  bf16* Vtb = (bf16*)(ws + ((size_t)32 << 20));     // 8 MiB  [B,H,HD,S]
  bf16* Ab  = (bf16*)(ws + ((size_t)40 << 20));     // 8 MiB  [4096,1024]

  cvt_f32_to_bf16<<<2048, 256, 0, stream>>>(hs, Xb, B_ * S_ * D_);
  cvt_transpose_bf16<<<2048, 256, 0, stream>>>(wa, Wat, D_, 3 * D_);
  cvt_transpose_bf16<<<1024, 256, 0, stream>>>(wp, Wpt, D_, D_);

  // QKV projection: [4096,1024] x [1024,3072], scatter epilogue
  gemm_wmma_bf16<<<dim3(3 * D_ / 128, B_ * S_ / 128), 256, 0, stream>>>(
      Xb, Wat, B_ * S_, 3 * D_, D_, 1, nullptr, Qb, Kb, Vtb);

  // attention (no mask, flash-style)
  attn_fwd<<<dim3(S_ / 128, H_, B_), 128, 0, stream>>>(Qb, Kb, Vtb, Ab);

  // output projection: [4096,1024] x [1024,1024] -> f32 d_out
  gemm_wmma_bf16<<<dim3(D_ / 128, B_ * S_ / 128), 256, 0, stream>>>(
      Ab, Wpt, B_ * S_, D_, D_, 0, out, nullptr, nullptr, nullptr);
}
